// DynamicAttention_41686952575666
// MI455X (gfx1250) — matrix operand order repaired and checked by reading, not compiled, not run on hardware
//
#include <hip/hip_runtime.h>
#include <hip/hip_bf16.h>
#include <cstdint>

// MI455X / gfx1250: all matmuls on v_wmma_f32_16x16x32_bf16, bf16
// intermediates, TDM (tensor_load_to_lds) staging of shared input panels,
// flash-softmax sliding-window attention with LDS P-tile transpose.

#define DEV __device__ __forceinline__

typedef __attribute__((ext_vector_type(16))) __bf16 v16bf;
typedef __attribute__((ext_vector_type(8)))  __bf16 v8bf;
typedef __attribute__((ext_vector_type(8)))  float  v8f;
typedef __attribute__((ext_vector_type(4)))  float  v4f;
typedef __attribute__((ext_vector_type(4)))  unsigned int u32x4;
typedef __attribute__((ext_vector_type(8)))  unsigned int u32x8;

constexpr int Bc   = 2;
constexpr int Sc   = 2048;
constexpr int Dc   = 1024;
constexpr int Hc   = 16;
constexpr int HDc  = 64;
constexpr int BLK  = 16;
constexpr int NBc  = Sc / BLK;        // 128 blocks per batch row of block_tables
constexpr int NTOK = Bc * Sc;         // 4096 tokens / slots

DEV v8f vzero8() {
    v8f z;
#pragma unroll
    for (int i = 0; i < 8; ++i) z[i] = 0.0f;
    return z;
}

DEV v16bf cat8(v8bf lo, v8bf hi) {
    v16bf r;
#pragma unroll
    for (int i = 0; i < 8; ++i) { r[i] = lo[i]; r[i + 8] = hi[i]; }
    return r;
}

DEV v8f wmma_bf16(v16bf a, v16bf b, v8f c) {
    // D = A(16x32 bf16) * B(32x16 bf16) + C(16x16 f32)
    return __builtin_amdgcn_wmma_f32_16x16x32_bf16(
        /*neg_a=*/false, a, /*neg_b=*/false, b,
        /*c_mod=*/(short)0, c, /*reuse_a=*/false, /*reuse_b=*/false);
}

// --------------------------------------------------------------- TDM panel
// DMA one 16 x 1024 bf16 panel (32 KiB) from global memory into LDS with the
// Tensor Data Mover. D# per CDNA5 ISA ch.8:
//  group0: count=1 | lds_addr | global_addr[56:0] | type=2
//  group1: data_size=2B, tensor_dim0=tile_dim0=1024, tensor_dim1=tile_dim1=16,
//          tensor_dim0_stride=1024, everything else 0 (2-D, no pad/iter/mask).
// Issued by one wave per block; completion via TENSORcnt.
DEV void tdm_load_panel_16xD(const __bf16* gsrc, __bf16* lds_panel) {
    const unsigned lds = (unsigned)(uintptr_t)lds_panel;           // LDS byte addr
    const unsigned long long ga = (unsigned long long)(uintptr_t)gsrc;
    u32x4 g0;
    g0[0] = 1u;                                   // count=1 (valid), no gather
    g0[1] = lds;                                  // lds_addr
    g0[2] = (unsigned)ga;                         // global_addr[31:0]
    g0[3] = (unsigned)(ga >> 32) | (2u << 30);    // global_addr[56:32] | type=2
    u32x8 g1;
    g1[0] = 0x00010000u;                          // data_size=1 -> 2 bytes
    g1[1] = (unsigned)Dc << 16;                   // tensor_dim0[15:0] @ bit48
    g1[2] = 16u << 16;                            // tensor_dim1[15:0] @ bit80
    g1[3] = (unsigned)Dc << 16;                   // tile_dim0 @ bit112
    g1[4] = 16u;                                  // tile_dim1 @ bit128
    g1[5] = (unsigned)Dc;                         // tensor_dim0_stride[31:0]
    g1[6] = 0u;
    g1[7] = 0u;
    // VADDR2/VADDR3 omitted (NULL) -> 2-D tensor form. Inline asm is the
    // toolchain-portable path (builtin arity differs across toolchains).
    asm volatile("tensor_load_to_lds %0, %1" :: "s"(g0), "s"(g1) : "memory");
}

// ---------------------------------------------------------------- convert
__global__ void k_f32_to_bf16(const float* __restrict__ src,
                              __bf16* __restrict__ dst, int n) {
    int i = blockIdx.x * blockDim.x + threadIdx.x;
    int stride = gridDim.x * blockDim.x;
    for (; i < n; i += stride) dst[i] = (__bf16)src[i];
}

// -------------------------------------------- Q/K projection (row-major out)
// Y[token][dim] = X @ W^T via C^T = W * X^T. Wave tile: 16 tokens x 64 dims.
// X panel staged to LDS by TDM (shared by all 8 waves of the block);
// A = W rows (lane = dim, global), B = X^T (lane = token, 32B ds load).
__global__ void k_proj_rowout(const __bf16* __restrict__ X,
                              const __bf16* __restrict__ W,
                              __bf16* __restrict__ Y,
                              const int* __restrict__ slot_map) {
    __shared__ __align__(128) __bf16 Xp[16 * Dc];   // 32 KiB panel

    const int lane = threadIdx.x & 31;
    const int hi = lane >> 4, ln = lane & 15;
    const int t0 = (blockIdx.x >> 1) * 16;                         // uniform
    const int ng = ((blockIdx.x & 1) << 3) + (threadIdx.x >> 5);   // 0..15
    const int n0 = ng * 64;

    if (threadIdx.x < 32) {
        tdm_load_panel_16xD(X + (size_t)t0 * Dc, Xp);
        __builtin_amdgcn_s_wait_tensorcnt(0);
    }
    __syncthreads();

    v8f acc[4];
#pragma unroll
    for (int g = 0; g < 4; ++g) acc[g] = vzero8();

    const __bf16* xl = Xp + ln * Dc;                 // B operand row (token)
    for (int k0 = 0; k0 < Dc; k0 += 32) {
        v16bf b = *(const v16bf*)(xl + k0 + hi * 16);
        if (k0 + 64 < Dc)
            __builtin_prefetch(W + (size_t)(n0 + ln) * Dc + k0 + 64, 0, 1);
#pragma unroll
        for (int g = 0; g < 4; ++g) {
            const __bf16* wrow = W + (size_t)(n0 + g * 16 + ln) * Dc + k0;
            v16bf a = *(const v16bf*)(wrow + hi * 16);
            acc[g] = wmma_bf16(a, b, acc[g]);
        }
    }

    const int t = t0 + ln;
    const int row = slot_map ? slot_map[t] : t;
#pragma unroll
    for (int g = 0; g < 4; ++g) {
        v8bf o;
#pragma unroll
        for (int r = 0; r < 8; ++r) o[r] = (__bf16)acc[g][r];
        *(v8bf*)(Y + (size_t)row * Dc + n0 + g * 16 + hi * 8) = o;
    }
}

// ------------------------------------------------- V projection (transposed)
// Vt[dim][slot] = (X @ W^T)^T. A = X rows from LDS panel (lane = token),
// B = W^T (lane = dim, 32B global load). C: lane = dim, tokens r+8*hi.
__global__ void k_proj_vT(const __bf16* __restrict__ X,
                          const __bf16* __restrict__ W,
                          __bf16* __restrict__ Vt,
                          const int* __restrict__ slot_map) {
    __shared__ __align__(128) __bf16 Xp[16 * Dc];

    const int lane = threadIdx.x & 31;
    const int hi = lane >> 4, ln = lane & 15;
    const int t0 = (blockIdx.x >> 1) * 16;
    const int ng = ((blockIdx.x & 1) << 3) + (threadIdx.x >> 5);
    const int n0 = ng * 64;

    if (threadIdx.x < 32) {
        tdm_load_panel_16xD(X + (size_t)t0 * Dc, Xp);
        __builtin_amdgcn_s_wait_tensorcnt(0);
    }
    __syncthreads();

    v8f acc[4];
#pragma unroll
    for (int g = 0; g < 4; ++g) acc[g] = vzero8();

    const __bf16* xl = Xp + ln * Dc;                 // A operand row (token)
    for (int k0 = 0; k0 < Dc; k0 += 32) {
        v16bf a = *(const v16bf*)(xl + k0 + hi * 16);
#pragma unroll
        for (int g = 0; g < 4; ++g) {
            const __bf16* wrow = W + (size_t)(n0 + g * 16 + ln) * Dc + k0;
            v16bf b = *(const v16bf*)(wrow + hi * 16);
            acc[g] = wmma_bf16(a, b, acc[g]);
        }
    }

#pragma unroll
    for (int g = 0; g < 4; ++g) {
        const size_t dim = (size_t)(n0 + g * 16 + ln);
#pragma unroll
        for (int r = 0; r < 8; ++r) {
            const int t = t0 + 8 * hi + r;
            const int slot = slot_map ? slot_map[t] : t;
            Vt[dim * NTOK + slot] = (__bf16)acc[g][r];
        }
    }
}

// ----------------------------------------------------------------- attention
// One wave per (batch, head, 16-query tile). Flash softmax over 32-key chunks.
// scores^T = K * Q^T  (C: lane = query, 8 keys per lane)
// out^T    = V^T * P  (P staged through LDS, ds_store_b128 / ds_load pair)
__global__ void k_attn(const __bf16* __restrict__ Q,
                       const __bf16* __restrict__ Kc,
                       const __bf16* __restrict__ Vt,
                       __bf16* __restrict__ AO,
                       const int* __restrict__ bt,
                       const int* __restrict__ ctx_lens,
                       const int* __restrict__ wsz) {
    __shared__ __align__(32) __bf16 Plds[4][16 * 48];   // row stride 48 (96B)

    const int w    = threadIdx.x >> 5;
    const int lane = threadIdx.x & 31;
    const int hi = lane >> 4, ln = lane & 15;
    const int wave = blockIdx.x * (blockDim.x >> 5) + w;

    const int QT = Sc / 16;
    const int qt = wave % QT;
    const int bh = wave / QT;
    const int h  = bh % Hc;
    const int b  = bh / Hc;
    const int q0 = qt * 16;

    const int win = wsz[0];
    const int ctx = ctx_lens[b];
    const int qg  = q0 + ln;                      // this lane's query position
    const float scale = 0.125f;                   // HD^-0.5, HD=64

    const __bf16* qbase = Q + (size_t)(b * Sc + q0 + ln) * Dc + h * HDc;
    __bf16* prow = &Plds[w][ln * 48];

    v8f acc[4];
#pragma unroll
    for (int g = 0; g < 4; ++g) acc[g] = vzero8();
    float m = -1e30f, lsum = 0.0f;

    int kmin = q0 - win + 1; if (kmin < 0) kmin = 0;
    const int kstart = (kmin >> 5) << 5;          // 32-aligned (= 2 blocks)
    const int kend   = q0 + 15;

    for (int kb = kstart; kb <= kend; kb += 32) {
        float p[16];
        float cmax = -1e30f;

        // ---- scores^T for two 16-key tiles of this chunk
#pragma unroll
        for (int t = 0; t < 2; ++t) {
            const int kt = kb + t * 16;
            const int sbase = bt[b * NBc + (kt >> 4)] * BLK;
            const __bf16* krow = Kc + (size_t)(sbase + ln) * Dc + h * HDc;
            v8f c = vzero8();
#pragma unroll
            for (int d0 = 0; d0 < HDc; d0 += 32) {
                v16bf a = *(const v16bf*)(krow + d0 + hi * 16);
                v16bf bb = *(const v16bf*)(qbase + d0 + hi * 16);
                c = wmma_bf16(a, bb, c);
            }
#pragma unroll
            for (int r = 0; r < 8; ++r) {
                const int key = kt + 8 * hi + r;
                float s = c[r] * scale;
                const bool vis = (key <= qg) && ((qg - key) < win) && (key < ctx);
                s = vis ? s : -1e30f;
                p[t * 8 + r] = s;
                cmax = fmaxf(cmax, s);
            }
        }

        // ---- online softmax update (lanes l and l^16 share a query)
        cmax = fmaxf(cmax, __shfl_xor(cmax, 16, 32));
        const float mnew  = fmaxf(m, cmax);
        const float alpha = __expf(m - mnew);
        float psum = 0.0f;
#pragma unroll
        for (int i = 0; i < 16; ++i) {
            const float pv = (p[i] <= -1e29f) ? 0.0f : __expf(p[i] - mnew);
            p[i] = pv;
            psum += pv;
        }
        psum += __shfl_xor(psum, 16, 32);
        lsum = lsum * alpha + psum;
        m = mnew;
#pragma unroll
        for (int g = 0; g < 4; ++g)
#pragma unroll
            for (int r = 0; r < 8; ++r) acc[g][r] *= alpha;

        // ---- stage P (bf16) through LDS: C-layout -> B-operand layout
#pragma unroll
        for (int t = 0; t < 2; ++t) {
            v8bf pp;
#pragma unroll
            for (int r = 0; r < 8; ++r) pp[r] = (__bf16)p[t * 8 + r];
            *(v8bf*)(prow + t * 16 + 8 * hi) = pp;
        }
        asm volatile("s_wait_dscnt 0" ::: "memory");   // same-wave LDS RAW
        v16bf pb = cat8(*(const v8bf*)(prow + hi * 8), *(const v8bf*)(prow + 16 + hi * 8));    // lane=query, 32 keys

        // ---- out^T += Vt * P
        const int sA = bt[b * NBc + (kb >> 4)] * BLK;
        const int sB = bt[b * NBc + (kb >> 4) + 1] * BLK;
#pragma unroll
        for (int g = 0; g < 4; ++g) {
            const __bf16* vrow = Vt + (size_t)(h * HDc + g * 16 + ln) * NTOK;
            v16bf a = cat8(*(const v8bf*)(vrow + sA + hi * 8),
                           *(const v8bf*)(vrow + sB + hi * 8));
            acc[g] = wmma_bf16(a, pb, acc[g]);
        }
    }

    const float inv = (lsum > 0.0f) ? (1.0f / lsum) : 0.0f;
    __bf16* orow = AO + (size_t)(b * Sc + q0 + ln) * Dc + h * HDc;
#pragma unroll
    for (int g = 0; g < 4; ++g) {
        v8bf o;
#pragma unroll
        for (int r = 0; r < 8; ++r) o[r] = (__bf16)(acc[g][r] * inv);
        *(v8bf*)(orow + g * 16 + hi * 8) = o;
    }
}

// ---------------------------------------------------------- output projection
// out[token][dim] (f32) = AO @ Wo^T; AO panel staged via TDM like k_proj_rowout.
__global__ void k_out_proj(const __bf16* __restrict__ AO,
                           const __bf16* __restrict__ W,
                           float* __restrict__ out) {
    __shared__ __align__(128) __bf16 Ap[16 * Dc];

    const int lane = threadIdx.x & 31;
    const int hi = lane >> 4, ln = lane & 15;
    const int t0 = (blockIdx.x >> 1) * 16;
    const int ng = ((blockIdx.x & 1) << 3) + (threadIdx.x >> 5);
    const int n0 = ng * 64;

    if (threadIdx.x < 32) {
        tdm_load_panel_16xD(AO + (size_t)t0 * Dc, Ap);
        __builtin_amdgcn_s_wait_tensorcnt(0);
    }
    __syncthreads();

    v8f acc[4];
#pragma unroll
    for (int g = 0; g < 4; ++g) acc[g] = vzero8();

    const __bf16* al = Ap + ln * Dc;
    for (int k0 = 0; k0 < Dc; k0 += 32) {
        v16bf b = *(const v16bf*)(al + k0 + hi * 16);
#pragma unroll
        for (int g = 0; g < 4; ++g) {
            const __bf16* wrow = W + (size_t)(n0 + g * 16 + ln) * Dc + k0;
            v16bf a = *(const v16bf*)(wrow + hi * 16);
            acc[g] = wmma_bf16(a, b, acc[g]);
        }
    }

    float* orow = out + (size_t)(t0 + ln) * Dc + n0;
#pragma unroll
    for (int g = 0; g < 4; ++g) {
        v4f o0, o1;
#pragma unroll
        for (int r = 0; r < 4; ++r) { o0[r] = acc[g][r]; o1[r] = acc[g][r + 4]; }
        *(v4f*)(orow + g * 16 + hi * 8)     = o0;
        *(v4f*)(orow + g * 16 + hi * 8 + 4) = o1;
    }
}

// ---------------------------------------------------------------------------
extern "C" void kernel_launch(void* const* d_in, const int* in_sizes, int n_in,
                              void* d_out, int out_size, void* d_ws, size_t ws_size,
                              hipStream_t stream) {
    (void)in_sizes; (void)n_in; (void)out_size; (void)ws_size;

    const float* x  = (const float*)d_in[0];
    const float* wq = (const float*)d_in[1];
    const float* wk = (const float*)d_in[2];
    const float* wv = (const float*)d_in[3];
    const float* wo = (const float*)d_in[4];
    const int* bt   = (const int*)d_in[5];
    const int* sm   = (const int*)d_in[6];
    const int* cl   = (const int*)d_in[7];
    const int* wsz  = (const int*)d_in[8];

    char* ws = (char*)d_ws;
    __bf16* xb  = (__bf16*)ws;  ws += (size_t)NTOK * Dc * 2;   // 8 MiB
    __bf16* wqb = (__bf16*)ws;  ws += (size_t)Dc * Dc * 2;     // 2 MiB
    __bf16* wkb = (__bf16*)ws;  ws += (size_t)Dc * Dc * 2;
    __bf16* wvb = (__bf16*)ws;  ws += (size_t)Dc * Dc * 2;
    __bf16* wob = (__bf16*)ws;  ws += (size_t)Dc * Dc * 2;
    __bf16* Qb  = (__bf16*)ws;  ws += (size_t)NTOK * Dc * 2;   // 8 MiB
    __bf16* Kcb = (__bf16*)ws;  ws += (size_t)NTOK * Dc * 2;   // 8 MiB (slot-major)
    __bf16* Vtb = (__bf16*)ws;  ws += (size_t)Dc * NTOK * 2;   // 8 MiB (dim-major)
    __bf16* AOb = (__bf16*)ws;  ws += (size_t)NTOK * Dc * 2;   // 8 MiB

    // fp32 -> bf16 conversions
    k_f32_to_bf16<<<2048, 256, 0, stream>>>(x,  xb,  NTOK * Dc);
    k_f32_to_bf16<<<1024, 256, 0, stream>>>(wq, wqb, Dc * Dc);
    k_f32_to_bf16<<<1024, 256, 0, stream>>>(wk, wkb, Dc * Dc);
    k_f32_to_bf16<<<1024, 256, 0, stream>>>(wv, wvb, Dc * Dc);
    k_f32_to_bf16<<<1024, 256, 0, stream>>>(wo, wob, Dc * Dc);

    // projections: 4096 waves each = 512 blocks x 8 waves
    k_proj_rowout<<<512, 256, 0, stream>>>(xb, wqb, Qb,  nullptr);  // Q[token][dim]
    k_proj_rowout<<<512, 256, 0, stream>>>(xb, wkb, Kcb, sm);       // K cache[slot][dim]
    k_proj_vT   <<<512, 256, 0, stream>>>(xb, wvb, Vtb, sm);        // V^T[dim][slot]

    // attention: B*H*(S/16) = 4096 waves = 1024 blocks x 4 waves
    k_attn<<<1024, 128, 0, stream>>>(Qb, Kcb, Vtb, AOb, bt, cl, wsz);

    // output projection -> f32 result
    k_out_proj<<<512, 256, 0, stream>>>(AOb, wob, (float*)d_out);
}
